// EpistaticMultiDecoder_33474975105568
// MI455X (gfx1250) — matrix operand order repaired and checked by reading, not compiled, not run on hardware
//
#include <hip/hip_runtime.h>

typedef __attribute__((ext_vector_type(16))) _Float16 v16h;
typedef __attribute__((ext_vector_type(8)))  _Float16 v8h;
typedef __attribute__((ext_vector_type(8)))  float    v8f;

#define MTOK 1000
#define JT   63                 // j tiles per i (63*16 = 1008 >= 1000)
#define NTILES (MTOK * JT)      // 63000 (i, j-tile) slots; ~half skipped by symmetry

__device__ __forceinline__ float half_red_add(float v) {
    // reduction within each 16-lane half of the wave32
    v += __shfl_xor(v, 1);
    v += __shfl_xor(v, 2);
    v += __shfl_xor(v, 4);
    v += __shfl_xor(v, 8);
    return v;
}

__device__ __forceinline__ v8f wmma16x16x32(v16h a, v16h b, v8f c) {
    // (neg_a, A, neg_b, B, c_mod, C, reuse_a, reuse_b)
    return __builtin_amdgcn_wmma_f32_16x16x32_f16(false, a, false, b, (short)0, c,
                                                  false, false);
}

// ---------------------------------------------------------------------------
// Kernel 1: adapter FFN + LayerNorm over 1000 tokens of dim 32.
// One wave per token, lane = feature dim. fm -> workspace (f32).
// ---------------------------------------------------------------------------
__global__ __launch_bounds__(256) void adapter_kernel(
    const float* __restrict__ feat,
    const float* __restrict__ aw1, const float* __restrict__ ab1,
    const float* __restrict__ aw2, const float* __restrict__ ab2,
    const float* __restrict__ ag,  const float* __restrict__ abt,
    float* __restrict__ fm)
{
    const int tok  = (blockIdx.x * blockDim.x + threadIdx.x) >> 5;
    const int lane = threadIdx.x & 31;
    if (tok >= MTOK) return;

    const float x = feat[tok * 32 + lane];

    float h = ab1[lane];
#pragma unroll
    for (int k = 0; k < 32; ++k)
        h += __shfl(x, k) * aw1[k * 32 + lane];
    h = fmaxf(h, 0.0f);

    float r = x + ab2[lane];
#pragma unroll
    for (int k = 0; k < 32; ++k)
        r += __shfl(h, k) * aw2[k * 32 + lane];

    float s = r;
#pragma unroll
    for (int m = 16; m >= 1; m >>= 1) s += __shfl_xor(s, m);
    const float mean = s * (1.0f / 32.0f);
    float d = r - mean;
    float v = d * d;
#pragma unroll
    for (int m = 16; m >= 1; m >>= 1) v += __shfl_xor(v, m);
    const float inv = rsqrtf(v * (1.0f / 32.0f) + 1e-5f);

    fm[tok * 32 + lane] = d * inv * ag[lane] + abt[lane];
}

// ---------------------------------------------------------------------------
// Kernel 2: fused pairwise head, upper-triangular tiles only (output is
// symmetric in (i,j); mirror-write both halves). Each wave: 16 pairs
// (i, j0..j0+15): LN -> WMMA GEMM -> ReLU -> WMMA GEMM -> residual -> LN -> dot.
// First-LN affine folded into W1/b1: GEMM1 consumes the *normalized* y'.
// ---------------------------------------------------------------------------
__global__ __launch_bounds__(256) void pair_head_kernel(
    const float* __restrict__ fm,  const float* __restrict__ ddg,
    const float* __restrict__ hg,  const float* __restrict__ hbt,
    const float* __restrict__ fw1, const float* __restrict__ fb1,
    const float* __restrict__ fw2, const float* __restrict__ fb2,
    const float* __restrict__ fg,  const float* __restrict__ fbt,
    const float* __restrict__ ow,  const float* __restrict__ ob,
    float* __restrict__ out)
{
    __shared__ __align__(64) _Float16 w1s[1024];      // diag(hg)*fw1, row-major f16
    __shared__ __align__(64) _Float16 w2s[1024];      // fw2 row-major f16
    __shared__ __align__(64) float    b1s[32];        // fb1 + hbt @ fw1
    __shared__ __align__(64) _Float16 ybuf[8][512];   // per-wave 16x32 f16 (y')
    __shared__ __align__(64) _Float16 hbuf[8][512];   // per-wave 16x32 f16 (C->A stage)

    const int tid  = threadIdx.x;
    const int lane = tid & 31;
    const int wv   = tid >> 5;

    for (int e = tid; e < 1024; e += 256) {
        w1s[e] = (_Float16)(hg[e >> 5] * fw1[e]);   // fold LN1 gamma into W1 rows
        w2s[e] = (_Float16)fw2[e];
    }
    if (tid < 32) {                                  // fold LN1 beta into b1
        float acc = fb1[tid];
#pragma unroll
        for (int k = 0; k < 32; ++k) acc += hbt[k] * fw1[k * 32 + tid];
        b1s[tid] = acc;
    }
    __syncthreads();

    // --- B fragments (lane%16 = N column, 16 K per lane in table order) ---
    v16h bw1_0, bw1_1, bw2_0, bw2_1;
    {
        const int bn = lane & 15, bh = lane >> 4;
#pragma unroll
        for (int s = 0; s < 16; ++s) {
            const int k = (s < 8) ? (8 * bh + s) : (16 + 8 * bh + (s - 8));
            bw1_0[s] = w1s[k * 32 + bn];
            bw1_1[s] = w1s[k * 32 + 16 + bn];
            bw2_0[s] = w2s[k * 32 + bn];
            bw2_1[s] = w2s[k * 32 + 16 + bn];
        }
    }

    // --- per-lane constants ---
    const int  row = lane & 15;            // A-matrix row / j offset owner
    const int  kb0 = (lane < 16) ? 0 : 16; // K half this lane owns pre-GEMM
    const int  Nl  = lane & 15;            // C-layout N index
    const bool lo  = (lane < 16);

    const float fb1n0 = b1s[Nl], fb1n1 = b1s[16 + Nl];
    const float hgn0  = hg[Nl],  hgn1  = hg[16 + Nl];
    // residual constant: fb2 + hbt at this lane's feature columns
    const float rc0 = fb2[Nl]      + hbt[Nl];
    const float rc1 = fb2[16 + Nl] + hbt[16 + Nl];
    // LayerNorm+dot folded: o = ir*(S3 - mean*Gsum) + Bsum
    const float gw0  = fg[Nl] * ow[Nl];
    const float gw1  = fg[16 + Nl] * ow[16 + Nl];
    const float Gsum = half_red_add(gw0 + gw1);
    const float Bsum = half_red_add(fbt[Nl] * ow[Nl] + fbt[16 + Nl] * ow[16 + Nl]);
    const float obv  = ob[0];

    for (int t = blockIdx.x * 8 + wv; t < NTILES; t += gridDim.x * 8) {
        const int i  = t / JT;
        const int j0 = (t - i * JT) * 16;
        if (j0 + 15 < i) continue;          // lower-triangular tile: mirrored elsewhere

        const int j  = j0 + row;
        const int jc = (j < MTOK) ? j : (MTOK - 1);

        // ---- x = fm[i] + fm[j] (16 features / lane), normalize (no affine) ----
        const float* pi = fm + i  * 32 + kb0;
        const float* pj = fm + jc * 32 + kb0;
        float x[16];
        float s = 0.0f;
#pragma unroll
        for (int u = 0; u < 16; ++u) { x[u] = pi[u] + pj[u]; s += x[u]; }
        s += __shfl_xor(s, 16);
        const float mean = s * (1.0f / 32.0f);
        float s2 = 0.0f;
#pragma unroll
        for (int u = 0; u < 16; ++u) { const float d = x[u] - mean; s2 += d * d; }
        s2 += __shfl_xor(s2, 16);
        const float inv = rsqrtf(s2 * (1.0f / 32.0f) + 1e-5f);

        v8h ylo, yhi;
#pragma unroll
        for (int u = 0; u < 8; ++u) {
            ylo[u] = (_Float16)((x[u]     - mean) * inv);
            yhi[u] = (_Float16)((x[u + 8] - mean) * inv);
        }
        {
            v8h* yp = (v8h*)(&ybuf[wv][row * 32 + kb0]);
            yp[0] = ylo; yp[1] = yhi;
        }
        asm volatile("s_wait_dscnt 0" ::: "memory");   // y' stores visible in-wave

        // ---- A fragment for GEMM1 (ISA 16-bit A 16x32 lane layout) ----
        v16h a1;
        {
            const v8h* pr = (const v8h*)(&ybuf[wv][row * 32]);
            const int  s0 = lo ? 0 : 1;
            ((v8h*)&a1)[0] = pr[s0];
            ((v8h*)&a1)[1] = pr[s0 + 2];
        }

        // ---- GEMM1: h1 = relu(y' @ W1' + b1') ----
        v8f c0 = {}, c1 = {};
        c0 = wmma16x16x32(a1, bw1_0, c0);
        c1 = wmma16x16x32(a1, bw1_1, c1);

#pragma unroll
        for (int r = 0; r < 8; ++r) {
            const int   Mr = lo ? r : (8 + r);
            const float h0 = fmaxf(c0[r] + fb1n0, 0.0f);
            const float h1 = fmaxf(c1[r] + fb1n1, 0.0f);
            hbuf[wv][Mr * 32 + Nl]      = (_Float16)h0;
            hbuf[wv][Mr * 32 + 16 + Nl] = (_Float16)h1;
        }
        asm volatile("s_wait_dscnt 0" ::: "memory");   // h1 stores visible in-wave

        // ---- A fragment for GEMM2 (C->A transpose via LDS) ----
        v16h a2;
        {
            const v8h* pr = (const v8h*)(&hbuf[wv][row * 32]);
            const int  s0 = lo ? 0 : 1;
            ((v8h*)&a2)[0] = pr[s0];
            ((v8h*)&a2)[1] = pr[s0 + 2];
        }

        // ---- GEMM2 + residual: z = (y'*hg + hbt) + h1 @ W2 + fb2 ----
        v8f z0 = {}, z1 = {};
        z0 = wmma16x16x32(a2, bw2_0, z0);
        z1 = wmma16x16x32(a2, bw2_1, z1);

#pragma unroll
        for (int r = 0; r < 8; ++r) {
            const int Mr = lo ? r : (8 + r);
            z0[r] += rc0 + hgn0 * (float)ybuf[wv][Mr * 32 + Nl];
            z1[r] += rc1 + hgn1 * (float)ybuf[wv][Mr * 32 + 16 + Nl];
        }

        // ---- folded final LN + dot + ddg adds, per C row ----
        const float cterm = Bsum + obv + ddg[i];
#pragma unroll
        for (int r = 0; r < 8; ++r) {
            const float S1 = half_red_add(z0[r] + z1[r]);
            const float S2 = half_red_add(z0[r] * z0[r] + z1[r] * z1[r]);
            const float S3 = half_red_add(z0[r] * gw0 + z1[r] * gw1);
            const float mr  = S1 * (1.0f / 32.0f);
            const float var = S2 * (1.0f / 32.0f) - mr * mr;
            const float ir  = rsqrtf(var + 1e-5f);
            const float o   = ir * (S3 - mr * Gsum) + cterm;
            if (Nl == 0) {
                const int jr = j0 + (lo ? r : (8 + r));
                if (jr < MTOK && jr >= i) {
                    const float val = o + ddg[jr];
                    out[i * MTOK + jr] = val;
                    if (jr != i) out[jr * MTOK + i] = val;
                }
            }
        }
    }
}

extern "C" void kernel_launch(void* const* d_in, const int* in_sizes, int n_in,
                              void* d_out, int out_size, void* d_ws, size_t ws_size,
                              hipStream_t stream) {
    (void)in_sizes; (void)n_in; (void)out_size; (void)ws_size;
    const float* feat = (const float*)d_in[0];
    const float* ddg  = (const float*)d_in[1];
    const float* aw1  = (const float*)d_in[2];
    const float* ab1  = (const float*)d_in[3];
    const float* aw2  = (const float*)d_in[4];
    const float* ab2  = (const float*)d_in[5];
    const float* ag   = (const float*)d_in[6];
    const float* abt  = (const float*)d_in[7];
    const float* hg   = (const float*)d_in[8];
    const float* hbt  = (const float*)d_in[9];
    const float* fw1  = (const float*)d_in[10];
    const float* fb1  = (const float*)d_in[11];
    const float* fw2  = (const float*)d_in[12];
    const float* fb2  = (const float*)d_in[13];
    const float* fg   = (const float*)d_in[14];
    const float* fbt  = (const float*)d_in[15];
    const float* ow   = (const float*)d_in[16];
    const float* ob   = (const float*)d_in[17];

    float* fm  = (float*)d_ws;    // 1000*32 f32 = 128 KB
    float* out = (float*)d_out;   // 1000*1000 f32

    adapter_kernel<<<125, 256, 0, stream>>>(feat, aw1, ab1, aw2, ab2, ag, abt, fm);

    pair_head_kernel<<<NTILES / 8, 256, 0, stream>>>(
        fm, ddg, hg, hbt, fw1, fb1, fw2, fb2, fg, fbt, ow, ob, out);
}
